// NeuralODEModel_71811853189738
// MI455X (gfx1250) — matrix operand order repaired and checked by reading, not compiled, not run on hardware
//
#include <hip/hip_runtime.h>
#include <math.h>

// ---------------------------------------------------------------------------
// Model dims
// ---------------------------------------------------------------------------
#define NN   64
#define KK   32
#define TT_  100            // encoder sequence length
#define FIN  2
#define DD   64
#define HH   4
#define DH   16
#define DFF  2048
#define NL   2
#define OUTD 2
#define NT   20             // ODE output times
#define SUB  8
#define BB   (NN*KK)        // 2048 sequences
#define BT   ((long)BB*TT_) // 204800 rows

typedef __attribute__((ext_vector_type(16))) __bf16 v16bf;
typedef __attribute__((ext_vector_type(8)))  __bf16 v8bf;
typedef __attribute__((ext_vector_type(8)))  float  v8f;

#define DEV static __device__ __forceinline__

// ---------------------------------------------------------------------------
// WMMA helpers (CDNA5 wave32, D = A(16x32 bf16) * B(32x16 bf16) + C(16x16 f32))
// ---------------------------------------------------------------------------
DEV v8f wmma_bf16(v16bf a, v16bf b, v8f c) {
  return __builtin_amdgcn_wmma_f32_16x16x32_bf16(false, a, false, b, (short)0, c,
                                                 false, false);
}

DEV v16bf cat8(v8bf lo, v8bf hi) {
  return __builtin_shufflevector(lo, hi, 0, 1, 2, 3, 4, 5, 6, 7,
                                 8, 9, 10, 11, 12, 13, 14, 15);
}

// A fragment 16x32 from bf16 row-major source (global or LDS), leading dim lda.
DEV v16bf fragA_bf16(const __bf16* base, int lda, int k0, int lane) {
  int m = lane & 15, half = lane >> 4;
  const __bf16* p = base + (long)m * lda + k0 + 8 * half;
  v8bf lo = *(const v8bf*)p;        // 16B -> b128
  v8bf hi = *(const v8bf*)(p + 16); // 16B -> b128
  return cat8(lo, hi);
}

// A fragment 16x32 from f32 row-major source (global), leading dim lda.
DEV v16bf fragA_f32(const float* A, long lda, long r0, int k0, int lane) {
  int m = lane & 15, half = lane >> 4;
  const float* p = A + (r0 + m) * lda + k0 + 8 * half;
  v8f lo = *(const v8f*)p;          // 32B
  v8f hi = *(const v8f*)(p + 16);   // 32B
  v16bf a;
#pragma unroll
  for (int i = 0; i < 8; ++i) {
    a[i]     = (__bf16)lo[i];
    a[8 + i] = (__bf16)hi[i];
  }
  return a;
}

// B fragment 32x16 from K-major (pre-transposed) bf16 weights WT[N][K].
DEV v16bf fragB_t(const __bf16* WT, int ldk, int k0, int c0, int lane) {
  int n = lane & 15, half = lane >> 4;
  const __bf16* p = WT + (long)(c0 + n) * ldk + k0 + 8 * half;
  v8bf lo = *(const v8bf*)p;
  v8bf hi = *(const v8bf*)(p + 16);
  return cat8(lo, hi);
}

// ---------------------------------------------------------------------------
// Kernel 0: weight preprocessing.  W[K,N] f32 -> WT[Npad,K] bf16.
// ---------------------------------------------------------------------------
__global__ void convT_kernel(const float* __restrict__ W, __bf16* __restrict__ WT,
                             int K, int N, int Npad) {
  int i = blockIdx.x * 256 + threadIdx.x;
  if (i >= Npad * K) return;
  int n = i / K, k = i - n * K;
  WT[i] = (n < N) ? (__bf16)W[(long)k * N + n] : (__bf16)0.0f;
}

// ---------------------------------------------------------------------------
// Kernel 1: embedding + positional encoding.
// ---------------------------------------------------------------------------
__global__ void embed_kernel(const float* __restrict__ x,
                             const float* __restrict__ ew,
                             const float* __restrict__ eb,
                             float* __restrict__ hout) {
  long i = (long)blockIdx.x * blockDim.x + threadIdx.x;
  if (i >= BT * (long)DD) return;
  long bt = i >> 6;
  int d = (int)(i & 63);
  int t = (int)(bt % TT_);
  float v = x[bt * 2] * ew[d] + x[bt * 2 + 1] * ew[DD + d] + eb[d];
  int j = d >> 1;
  float freq = __expf(-(float)(2 * j) * 0.14391156831f);  // ln(10000)/64
  float ang = (float)t * freq;
  v += (d & 1) ? __cosf(ang) : __sinf(ang);
  hout[i] = v;
}

// ---------------------------------------------------------------------------
// Kernel 2: fused MHA core, one block per (batch, head).
// ---------------------------------------------------------------------------
__global__ void attn_kernel(const float* __restrict__ h,
                            const float* __restrict__ wq, const float* __restrict__ bq,
                            const float* __restrict__ wk, const float* __restrict__ bk,
                            const float* __restrict__ wv, const float* __restrict__ bv,
                            float* __restrict__ attn_out) {
  __shared__ float hs[TT_ * DD];
  __shared__ float qs[TT_ * DH], ks_[TT_ * DH], vs[TT_ * DH];
  int b = blockIdx.x >> 2;
  int hh = blockIdx.x & 3;
  int tid = threadIdx.x;  // 128 threads
  const float* hb = h + (long)b * TT_ * DD;
  for (int i = tid; i < TT_ * DD; i += 128) hs[i] = hb[i];
  __syncthreads();
  for (int i = tid; i < TT_ * DH; i += 128) {
    int t = i >> 4, d = i & 15;
    int col = hh * DH + d;
    float sq = bq[col], sk = bk[col], sv = bv[col];
    for (int c = 0; c < DD; ++c) {
      float hv = hs[t * DD + c];
      sq += hv * wq[c * DD + col];
      sk += hv * wk[c * DD + col];
      sv += hv * wv[c * DD + col];
    }
    qs[i] = sq; ks_[i] = sk; vs[i] = sv;
  }
  __syncthreads();
  if (tid < TT_) {
    float s[TT_];
    float mx = -1e30f;
    for (int j = 0; j < TT_; ++j) {
      float d = 0.f;
      for (int c = 0; c < DH; ++c) d += qs[tid * DH + c] * ks_[j * DH + c];
      d *= 0.25f;  // 1/sqrt(16)
      s[j] = d;
      mx = fmaxf(mx, d);
    }
    float sum = 0.f;
    for (int j = 0; j < TT_; ++j) { float e = __expf(s[j] - mx); s[j] = e; sum += e; }
    float inv = 1.f / sum;
    float o[DH];
    for (int c = 0; c < DH; ++c) o[c] = 0.f;
    for (int j = 0; j < TT_; ++j) {
      float p = s[j] * inv;
      for (int c = 0; c < DH; ++c) o[c] += p * vs[j * DH + c];
    }
    for (int c = 0; c < DH; ++c)
      attn_out[(long)b * TT_ * DD + tid * DD + hh * DH + c] = o[c];
  }
}

// ---------------------------------------------------------------------------
// Kernel 3: WMMA GEMM  C[M,N] = A[M,K] @ WT^T + bias  (WT is [N,K] bf16).
// ---------------------------------------------------------------------------
__global__ void gemm_wmma_kernel(const float* __restrict__ A,
                                 const __bf16* __restrict__ WT,
                                 const float* __restrict__ bias,
                                 float* __restrict__ C,
                                 long M, int N, int Kd) {
  int lane = threadIdx.x & 31;
  long mt = (long)blockIdx.x * 16;
  int nt = blockIdx.y * 16;
  v8f acc = {};
  for (int k0 = 0; k0 < Kd; k0 += 32) {
    v16bf a = fragA_f32(A, Kd, mt, k0, lane);
    v16bf b = fragB_t(WT, Kd, k0, nt, lane);
    acc = wmma_bf16(a, b, acc);
  }
  int n = nt + (lane & 15);
  float bv = bias[n];
#pragma unroll
  for (int r = 0; r < 8; ++r) {
    long m = mt + r + 8 * (lane >> 4);
    C[m * (long)N + n] = acc[r] + bv;
  }
}

// ---------------------------------------------------------------------------
// Kernel 4: residual + LayerNorm per row (64 features)
// ---------------------------------------------------------------------------
__global__ void ln_kernel(const float* __restrict__ X, const float* __restrict__ R,
                          const float* __restrict__ g, const float* __restrict__ bb,
                          float* __restrict__ Y, long M) {
  long r = (long)blockIdx.x * blockDim.x + threadIdx.x;
  if (r >= M) return;
  float v[DD];
  float mean = 0.f;
  for (int d = 0; d < DD; ++d) {
    float t = X[r * DD + d] + R[r * DD + d];
    v[d] = t;
    mean += t;
  }
  mean *= (1.f / DD);
  float var = 0.f;
  for (int d = 0; d < DD; ++d) { float dd = v[d] - mean; var += dd * dd; }
  var *= (1.f / DD);
  float inv = rsqrtf(var + 1e-5f);
  for (int d = 0; d < DD; ++d)
    Y[r * DD + d] = (v[d] - mean) * inv * g[d] + bb[d];
}

// ---------------------------------------------------------------------------
// Kernel 5: fused FFN, 2 M-tiles (32 rows) per wave so every B fragment feeds
// two WMMAs.  Hidden dim in chunks of 32 staged through LDS; next chunk of
// weights prefetched (global_prefetch_b8).
// ---------------------------------------------------------------------------
__global__ void ffn_fused_kernel(const float* __restrict__ X,
                                 const __bf16* __restrict__ w1T,  // [2048,64]
                                 const float* __restrict__ b1,
                                 const __bf16* __restrict__ w2T,  // [64,2048]
                                 const float* __restrict__ b2,
                                 float* __restrict__ Y, long M) {
  __shared__ __attribute__((aligned(32))) __bf16 tst[32 * 32];
  int lane = threadIdx.x & 31;
  long mt = (long)blockIdx.x * 32;
  int nloc = lane & 15, half = lane >> 4;
  v16bf a0 = fragA_f32(X, DD, mt, 0, lane);
  v16bf a1 = fragA_f32(X, DD, mt, 32, lane);
  v16bf a2 = fragA_f32(X, DD, mt + 16, 0, lane);
  v16bf a3 = fragA_f32(X, DD, mt + 16, 32, lane);
  v8f accY0[4], accY1[4];
#pragma unroll
  for (int jn = 0; jn < 4; ++jn) { accY0[jn] = (v8f){}; accY1[jn] = (v8f){}; }
  for (int j = 0; j < DFF / 32; ++j) {
    int jj = j * 32;
    if (j + 1 < DFF / 32) {  // cover L2 latency for next chunk's weights
      __builtin_prefetch(w1T + (long)(jj + 32) * DD, 0, 1);
      __builtin_prefetch(w2T + jj + 32, 0, 1);
    }
#pragma unroll
    for (int tile = 0; tile < 2; ++tile) {
      int c0 = jj + tile * 16;
      v16bf b0 = fragB_t(w1T, DD, 0, c0, lane);
      v16bf bv1 = fragB_t(w1T, DD, 32, c0, lane);
      v8f acc0 = {}, acc1 = {};
      acc0 = wmma_bf16(a0, b0, acc0);
      acc0 = wmma_bf16(a1, bv1, acc0);
      acc1 = wmma_bf16(a2, b0, acc1);
      acc1 = wmma_bf16(a3, bv1, acc1);
      float bv = b1[c0 + nloc];
#pragma unroll
      for (int r = 0; r < 8; ++r) {
        int m = r + 8 * half;
        tst[m * 32 + tile * 16 + nloc]        = (__bf16)fmaxf(acc0[r] + bv, 0.f);
        tst[(16 + m) * 32 + tile * 16 + nloc] = (__bf16)fmaxf(acc1[r] + bv, 0.f);
      }
    }
    __syncthreads();
    v16bf at0 = fragA_bf16(tst, 32, 0, lane);
    v16bf at1 = fragA_bf16(tst + 16 * 32, 32, 0, lane);
#pragma unroll
    for (int jn = 0; jn < 4; ++jn) {
      v16bf b2f = fragB_t(w2T, DFF, jj, jn * 16, lane);
      accY0[jn] = wmma_bf16(at0, b2f, accY0[jn]);
      accY1[jn] = wmma_bf16(at1, b2f, accY1[jn]);
    }
    __syncthreads();
  }
#pragma unroll
  for (int jn = 0; jn < 4; ++jn) {
    int n = jn * 16 + nloc;
    float bv = b2[n];
#pragma unroll
    for (int r = 0; r < 8; ++r) {
      long m = mt + r + 8 * half;
      Y[m * DD + n]        = accY0[jn][r] + bv;
      Y[(m + 16) * DD + n] = accY1[jn][r] + bv;
    }
  }
}

// ---------------------------------------------------------------------------
// Kernel 6: fully fused Dopri5 ODE integration, 4 cooperating waves per block.
// Each wave owns one 16-col n-tile of z / k1..k6 (one v8f each); GEMM1's 8
// hidden tiles are split 2 per wave; all exchange through zst/tst in LDS.
// 128 blocks x 4 waves = 512 waves during the sequential phase.
// ---------------------------------------------------------------------------
DEV void stageStore4(__bf16* zst, v8f z, int lane, int wid) {
  int nloc = lane & 15, half = lane >> 4;
#pragma unroll
  for (int r = 0; r < 8; ++r)
    zst[(r + 8 * half) * DD + wid * 16 + nloc] = (__bf16)z[r];
  __syncthreads();
}

DEV v8f fodev4(const __bf16* zst, __bf16* tst,
               const __bf16* w1Ts, const __bf16* w2Ts,
               const float* b1s, const float* b2s, int lane, int wid) {
  int nloc = lane & 15, half = lane >> 4;
  v16bf a0 = fragA_bf16(zst, DD, 0, lane);
  v16bf a1 = fragA_bf16(zst, DD, 32, lane);
#pragma unroll
  for (int jj = 0; jj < 2; ++jj) {
    int c0 = (2 * wid + jj) * 16;
    v8f acc = {};
    acc = wmma_bf16(a0, fragB_t(w1Ts, DD, 0, c0, lane), acc);
    acc = wmma_bf16(a1, fragB_t(w1Ts, DD, 32, c0, lane), acc);
    float bv = b1s[c0 + nloc];
#pragma unroll
    for (int r = 0; r < 8; ++r)
      tst[(r + 8 * half) * 128 + c0 + nloc] = (__bf16)tanhf(acc[r] + bv);
  }
  __syncthreads();
  v8f acc = {};
#pragma unroll
  for (int ks = 0; ks < 4; ++ks) {
    v16bf at = fragA_bf16(tst, 128, ks * 32, lane);
    acc = wmma_bf16(at, fragB_t(w2Ts, 128, ks * 32, wid * 16, lane), acc);
  }
  float bv = b2s[wid * 16 + nloc];
#pragma unroll
  for (int r = 0; r < 8; ++r) acc[r] += bv;
  __syncthreads();
  return acc;
}

__global__ void ode_kernel(const float* __restrict__ h0,
                           const float* __restrict__ tarr,
                           const __bf16* __restrict__ w1T,  // [128,64] bf16
                           const float* __restrict__ b1,
                           const __bf16* __restrict__ w2T,  // [64,128] bf16
                           const float* __restrict__ b2,
                           float* __restrict__ zt) {
  __shared__ __attribute__((aligned(32))) __bf16 w1Ts[128 * DD];
  __shared__ __attribute__((aligned(32))) __bf16 w2Ts[DD * 128];
  __shared__ __attribute__((aligned(32))) __bf16 zst[16 * DD];
  __shared__ __attribute__((aligned(32))) __bf16 tst[16 * 128];
  __shared__ float b1s[128], b2s[DD];
  int tid = threadIdx.x, lane = tid & 31, wid = tid >> 5;  // 128 threads, 4 waves
  {
    const unsigned* s1 = (const unsigned*)w1T;
    const unsigned* s2 = (const unsigned*)w2T;
    unsigned* d1 = (unsigned*)w1Ts;
    unsigned* d2 = (unsigned*)w2Ts;
    for (int i = tid; i < 128 * DD / 2; i += 128) { d1[i] = s1[i]; d2[i] = s2[i]; }
    for (int i = tid; i < 128; i += 128) b1s[i] = b1[i];
    if (tid < DD) b2s[tid] = b2[tid];
  }
  __syncthreads();

  long g0 = (long)blockIdx.x * 16;
  int nloc = lane & 15, half = lane >> 4;
  int c = wid * 16 + nloc;  // this wave's column slice

  const float A21 = 1.f / 5.f;
  const float A31 = 3.f / 40.f, A32 = 9.f / 40.f;
  const float A41 = 44.f / 45.f, A42 = -56.f / 15.f, A43 = 32.f / 9.f;
  const float A51 = 19372.f / 6561.f, A52 = -25360.f / 2187.f,
              A53 = 64448.f / 6561.f, A54 = -212.f / 729.f;
  const float A61 = 9017.f / 3168.f, A62 = -355.f / 33.f, A63 = 46732.f / 5247.f,
              A64 = 49.f / 176.f, A65 = -5103.f / 18656.f;
  const float B1 = 35.f / 384.f, B3 = 500.f / 1113.f, B4 = 125.f / 192.f,
              B5 = -2187.f / 6784.f, B6 = 11.f / 84.f;

  v8f zc;
#pragma unroll
  for (int r = 0; r < 8; ++r)
    zc[r] = h0[(g0 + r + 8 * half) * (long)(TT_ * DD) + c];

#pragma unroll
  for (int r = 0; r < 8; ++r)
    zt[(g0 + r + 8 * half) * (long)(NT * DD) + c] = zc[r];

  for (int it = 0; it < NT - 1; ++it) {
    float hstep = (tarr[it + 1] - tarr[it]) * (1.f / SUB);
    for (int s = 0; s < SUB; ++s) {
      v8f k1, k2, k3, k4, k5, k6, st;
      stageStore4(zst, zc, lane, wid);
      k1 = fodev4(zst, tst, w1Ts, w2Ts, b1s, b2s, lane, wid);
      st = zc + hstep * (A21 * k1);
      stageStore4(zst, st, lane, wid);
      k2 = fodev4(zst, tst, w1Ts, w2Ts, b1s, b2s, lane, wid);
      st = zc + hstep * (A31 * k1 + A32 * k2);
      stageStore4(zst, st, lane, wid);
      k3 = fodev4(zst, tst, w1Ts, w2Ts, b1s, b2s, lane, wid);
      st = zc + hstep * (A41 * k1 + A42 * k2 + A43 * k3);
      stageStore4(zst, st, lane, wid);
      k4 = fodev4(zst, tst, w1Ts, w2Ts, b1s, b2s, lane, wid);
      st = zc + hstep * (A51 * k1 + A52 * k2 + A53 * k3 + A54 * k4);
      stageStore4(zst, st, lane, wid);
      k5 = fodev4(zst, tst, w1Ts, w2Ts, b1s, b2s, lane, wid);
      st = zc + hstep * (A61 * k1 + A62 * k2 + A63 * k3 + A64 * k4 + A65 * k5);
      stageStore4(zst, st, lane, wid);
      k6 = fodev4(zst, tst, w1Ts, w2Ts, b1s, b2s, lane, wid);
      zc = zc + hstep * (B1 * k1 + B3 * k3 + B4 * k4 + B5 * k5 + B6 * k6);
    }
#pragma unroll
    for (int r = 0; r < 8; ++r)
      zt[(g0 + r + 8 * half) * (long)(NT * DD) + (it + 1) * DD + c] = zc[r];
  }
}

// ---------------------------------------------------------------------------
// Kernel 7: fused decoder  out = tanh(z@dw1+db1)@dw2+db2, one wave / 16 rows.
// dw2T is [16,128] bf16 with rows n>=2 zero-padded.
// ---------------------------------------------------------------------------
__global__ void dec_kernel(const float* __restrict__ Z,
                           const __bf16* __restrict__ w1T,  // [128,64]
                           const float* __restrict__ b1,
                           const __bf16* __restrict__ w2T,  // [16,128] padded
                           const float* __restrict__ b2,
                           float* __restrict__ out, long M) {
  __shared__ __attribute__((aligned(32))) __bf16 tst[16 * 128];
  int lane = threadIdx.x & 31;
  long mt = (long)blockIdx.x * 16;
  int nloc = lane & 15, half = lane >> 4;
  v16bf a0 = fragA_f32(Z, DD, mt, 0, lane);
  v16bf a1 = fragA_f32(Z, DD, mt, 32, lane);
  for (int j = 0; j < 8; ++j) {
    int c0 = j * 16;
    v8f acc = {};
    acc = wmma_bf16(a0, fragB_t(w1T, DD, 0, c0, lane), acc);
    acc = wmma_bf16(a1, fragB_t(w1T, DD, 32, c0, lane), acc);
    float bv = b1[c0 + nloc];
#pragma unroll
    for (int r = 0; r < 8; ++r)
      tst[(r + 8 * half) * 128 + c0 + nloc] = (__bf16)tanhf(acc[r] + bv);
  }
  __syncthreads();
  v8f acc = {};
  for (int ks = 0; ks < 4; ++ks) {
    v16bf at = fragA_bf16(tst, 128, ks * 32, lane);
    acc = wmma_bf16(at, fragB_t(w2T, 128, ks * 32, 0, lane), acc);
  }
  if (nloc < OUTD) {
    float bv = b2[nloc];
#pragma unroll
    for (int r = 0; r < 8; ++r) {
      long m = mt + r + 8 * half;
      out[m * OUTD + nloc] = acc[r] + bv;
    }
  }
}

// ---------------------------------------------------------------------------
// Host side
// ---------------------------------------------------------------------------
struct LayerP {
  const float *wq, *bq, *wk, *bk, *wv, *bv, *wo, *bo;
  const float *ln1g, *ln1b, *ln2g, *ln2b;
  const float *ffw1, *ffb1, *ffw2, *ffb2;
};

static inline void launch_convT(const float* W, __bf16* WT, int K, int N, int Npad,
                                hipStream_t stream) {
  int total = Npad * K;
  convT_kernel<<<(total + 255) / 256, 256, 0, stream>>>(W, WT, K, N, Npad);
}

extern "C" void kernel_launch(void* const* d_in, const int* in_sizes, int n_in,
                              void* d_out, int out_size, void* d_ws, size_t ws_size,
                              hipStream_t stream) {
  const float* x; const float* tvec; const float *emb_w, *emb_b;
  LayerP L[NL];
  const float *ow1, *ob1, *ow2, *ob2, *dw1, *db1, *dw2, *db2;

  auto F = [&](int i) { return (const float*)d_in[i]; };

  if (in_sizes[0] == NN * KK * TT_ * FIN) {
    // setup_inputs() insertion order: x, t, emb_w, emb_b, layers(16 each), ode*, dec*
    x = F(0); tvec = F(1); emb_w = F(2); emb_b = F(3);
    int p = 4;
    for (int l = 0; l < NL; ++l) {
      L[l].wq = F(p + 0);  L[l].bq = F(p + 1);
      L[l].wk = F(p + 2);  L[l].bk = F(p + 3);
      L[l].wv = F(p + 4);  L[l].bv = F(p + 5);
      L[l].wo = F(p + 6);  L[l].bo = F(p + 7);
      L[l].ln1g = F(p + 8);  L[l].ln1b = F(p + 9);
      L[l].ln2g = F(p + 10); L[l].ln2b = F(p + 11);
      L[l].ffw1 = F(p + 12); L[l].ffb1 = F(p + 13);
      L[l].ffw2 = F(p + 14); L[l].ffb2 = F(p + 15);
      p += 16;
    }
    ow1 = F(p + 0); ob1 = F(p + 1); ow2 = F(p + 2); ob2 = F(p + 3);
    dw1 = F(p + 4); db1 = F(p + 5); dw2 = F(p + 6); db2 = F(p + 7);
  } else {
    // alphabetical pytree-leaf order
    db1 = F(0); db2 = F(1); dw1 = F(2); dw2 = F(3);
    emb_b = F(4); emb_w = F(5);
    int p = 6;
    for (int l = 0; l < NL; ++l) {
      L[l].bk = F(p + 0);  L[l].bo = F(p + 1);
      L[l].bq = F(p + 2);  L[l].bv = F(p + 3);
      L[l].ffb1 = F(p + 4); L[l].ffb2 = F(p + 5);
      L[l].ffw1 = F(p + 6); L[l].ffw2 = F(p + 7);
      L[l].ln1b = F(p + 8);  L[l].ln1g = F(p + 9);
      L[l].ln2b = F(p + 10); L[l].ln2g = F(p + 11);
      L[l].wk = F(p + 12); L[l].wo = F(p + 13);
      L[l].wq = F(p + 14); L[l].wv = F(p + 15);
      p += 16;
    }
    ob1 = F(p + 0); ob2 = F(p + 1); ow1 = F(p + 2); ow2 = F(p + 3);
    tvec = F(p + 4); x = F(p + 5);
  }

  // workspace layout: h | buf1 | buf2 | zt (f32)  then bf16 transposed weights
  const long ACT = BT * (long)DD;  // 13,107,200 floats
  float* h    = (float*)d_ws;
  float* buf1 = h + ACT;
  float* buf2 = buf1 + ACT;
  float* zt   = buf2 + ACT;                       // [BB, NT, DD]
  __bf16* bfw = (__bf16*)(zt + (long)BB * NT * DD);
  __bf16* woT[NL], *w1T[NL], *w2T[NL];
  for (int l = 0; l < NL; ++l) {
    woT[l] = bfw; bfw += DD * DD;        // [64,64]
    w1T[l] = bfw; bfw += DFF * DD;       // [2048,64]
    w2T[l] = bfw; bfw += DD * DFF;       // [64,2048]
  }
  __bf16* ow1T = bfw; bfw += 128 * DD;   // [128,64]
  __bf16* ow2T = bfw; bfw += DD * 128;   // [64,128]
  __bf16* dw1T = bfw; bfw += 128 * DD;   // [128,64]
  __bf16* dw2T = bfw; bfw += 16 * 128;   // [16,128] zero-padded

  // 0) preprocess weights: transpose + f32->bf16 (+ zero-pad decoder head)
  for (int l = 0; l < NL; ++l) {
    launch_convT(L[l].wo,  woT[l], DD, DD, DD, stream);
    launch_convT(L[l].ffw1, w1T[l], DD, DFF, DFF, stream);
    launch_convT(L[l].ffw2, w2T[l], DFF, DD, DD, stream);
  }
  launch_convT(ow1, ow1T, DD, 128, 128, stream);
  launch_convT(ow2, ow2T, 128, DD, DD, stream);
  launch_convT(dw1, dw1T, DD, 128, 128, stream);
  launch_convT(dw2, dw2T, 128, OUTD, 16, stream);

  // 1) embedding + positional encoding
  embed_kernel<<<(unsigned)((ACT + 255) / 256), 256, 0, stream>>>(x, emb_w, emb_b, h);

  // 2) encoder layers
  for (int l = 0; l < NL; ++l) {
    attn_kernel<<<BB * HH, 128, 0, stream>>>(h, L[l].wq, L[l].bq, L[l].wk, L[l].bk,
                                             L[l].wv, L[l].bv, buf1);
    {
      dim3 grid((unsigned)(BT / 16), DD / 16);
      gemm_wmma_kernel<<<grid, 32, 0, stream>>>(buf1, woT[l], L[l].bo, buf2,
                                                BT, DD, DD);
    }
    ln_kernel<<<(unsigned)((BT + 255) / 256), 256, 0, stream>>>(h, buf2, L[l].ln1g,
                                                                L[l].ln1b, h, BT);
    ffn_fused_kernel<<<(unsigned)(BT / 32), 32, 0, stream>>>(h, w1T[l], L[l].ffb1,
                                                             w2T[l], L[l].ffb2,
                                                             buf1, BT);
    ln_kernel<<<(unsigned)((BT + 255) / 256), 256, 0, stream>>>(h, buf1, L[l].ln2g,
                                                                L[l].ln2b, h, BT);
  }

  // 3) fused Dopri5 ODE integration (4 waves/block): zt [BB, NT, DD]
  ode_kernel<<<BB / 16, 128, 0, stream>>>(h, tvec, ow1T, ob1, ow2T, ob2, zt);

  // 4) fused decoder: out[BB*NT, OUTD]
  {
    long Mdec = (long)BB * NT;  // 40960
    dec_kernel<<<(unsigned)(Mdec / 16), 32, 0, stream>>>(zt, dw1T, db1, dw2T, db2,
                                                         (float*)d_out, Mdec);
  }
}
